// TransformerBlock_82300163325971
// MI455X (gfx1250) — compile-verified, matrix operand order checked by reading, not run on hardware
//
#include <hip/hip_runtime.h>
#include <hip/hip_bf16.h>
#include <math.h>

#define DEV static __device__ __forceinline__

typedef __bf16 bf16;
typedef __attribute__((ext_vector_type(16))) __bf16 bf16x16;
typedef __attribute__((ext_vector_type(8)))  __bf16 bf16x8;
typedef __attribute__((ext_vector_type(8)))  float  f32x8;

// ---------- helpers ----------

DEV bf16 f2bf(float f) {
  union { float f; unsigned u; } un; un.f = f;
  unsigned r = un.u + 0x7FFFu + ((un.u >> 16) & 1u);   // round-to-nearest-even
  unsigned short h = (unsigned short)(r >> 16);
  bf16 out; __builtin_memcpy(&out, &h, 2); return out;
}

DEV f32x8 wmma_bf16(bf16x16 a, bf16x16 b, f32x8 c) {
  // D(16x16 f32) = A(16x32 bf16) * B(32x16 bf16) + C
  return __builtin_amdgcn_wmma_f32_16x16x32_bf16(false, a, false, b, (short)0, c,
                                                 false, false);
}

// Fragment loader for A (row = M index) or B-from-transposed (row = N index).
// Row-major matrix with leading dim `ld` (elements); loads the 16 bf16 this
// lane owns for a 16x32(K) fragment starting at column k0.
// Per ISA layout: lane half h owns K = {h*8..h*8+7} U {16+h*8..16+h*8+7}.
DEV bf16x16 load_frag_rows(const bf16* __restrict__ base, int ld, int row,
                           int k0, int half) {
  const bf16* p = base + (size_t)row * ld + k0 + half * 8;
  bf16x8 lo = *(const bf16x8*)(p);        // 16B contiguous
  bf16x8 hi = *(const bf16x8*)(p + 16);   // 16B contiguous
  bf16x16 f;
#pragma unroll
  for (int i = 0; i < 8; ++i) { f[i] = lo[i]; f[i + 8] = hi[i]; }
  return f;
}

// Async copy of 16 bytes global -> LDS (gfx1250 async path, ASYNCcnt tracked).
DEV void async_g2l_b128(uint32_t lds_byte_off, const bf16* gptr) {
  asm volatile("global_load_async_to_lds_b128 %0, %1, off"
               :: "v"(lds_byte_off), "v"(gptr) : "memory");
}

// ---------- conversion kernels ----------

__global__ void cvt_bf16_kernel(const float* __restrict__ in, bf16* __restrict__ out,
                                long n) {
  long i = (long)blockIdx.x * blockDim.x + threadIdx.x;
  if (i < n) out[i] = f2bf(in[i]);
}

// src is RxC fp32 row-major; dst is CxR bf16 row-major (transpose).
__global__ void cvt_bf16_T_kernel(const float* __restrict__ in, bf16* __restrict__ out,
                                  int R, int C) {
  long i = (long)blockIdx.x * blockDim.x + threadIdx.x;
  long n = (long)R * C;
  if (i < n) {
    int r = (int)(i / C), c = (int)(i % C);
    out[(size_t)c * R + r] = f2bf(in[i]);
  }
}

// ---------- GEMM: C(MxN) = A(MxK) @ B(KxN) + bias, B given transposed (NxK) ----
// Block = 256 threads (8 waves) computing a 128x128 tile, k-step 64.
// Double-buffered LDS stages (async-to-LDS); per k-step each wave does 16 WMMAs
// on its 32x64 sub-tile. Async completion is in-order, so waiting ASYNCcnt<=8
// after issuing the next stage's 8 transfers guarantees the previous stage
// has landed.
// MODE 0: store bf16(acc+bias)                      -> outB[row*N+col]
// MODE 1: store fp32(acc+bias+residual)             -> outF[row*N+col]
// MODE 2: store bf16(gelu_exact(acc+bias))          -> outB[row*N+col]
// MODE 3: store bf16(acc+bias) transposed per batch -> outB[(b*N+col)*S + s]
template <int MODE>
__global__ void gemm_bf16_kernel(const bf16* __restrict__ A, const bf16* __restrict__ BT,
                                 const float* __restrict__ bias,
                                 const float* __restrict__ residual,
                                 float* __restrict__ outF, bf16* __restrict__ outB,
                                 int M, int N, int K, int S) {
  __shared__ __align__(16) bf16 tA[2][128 * 64];   // 2 x 16 KB
  __shared__ __align__(16) bf16 tB[2][128 * 64];   // 2 x 16 KB

  int tid  = threadIdx.x;
  int lane = tid & 31;
  int half = lane >> 4;
  int mn   = lane & 15;
  int w    = tid >> 5;
  int wm   = w & 3;        // wave M sub-tile (32 rows of 128)
  int wn   = w >> 2;       // wave N sub-tile (64 cols of 128)

  int blocks_n = N >> 7;
  int bm = (int)blockIdx.x / blocks_n;
  int bn = (int)blockIdx.x % blocks_n;
  int mbase = bm * 128;
  int nbase = bn * 128;

  // staging: tile row has 64 bf16 = 128 B = 8 x 16B chunks.
  // thread handles chunk (tid & 7) of rows (tid>>3) + 32*p, p = 0..3.
  int srow = tid >> 3;            // 0..31
  int schk = tid & 7;             // 0..7
  uint32_t lA0 = (uint32_t)(uintptr_t)&tA[0][0] + (uint32_t)(srow * 128 + schk * 16);
  uint32_t lB0 = (uint32_t)(uintptr_t)&tB[0][0] + (uint32_t)(srow * 128 + schk * 16);
  uint32_t bufStride = (uint32_t)(128 * 64 * 2);   // bytes between buffers
  uint32_t rowStep   = 32 * 128;                   // 32 rows in bytes

  const bf16* gA = A  + (size_t)(mbase + srow) * K + schk * 8;
  const bf16* gB = BT + (size_t)(nbase + srow) * K + schk * 8;
  size_t gRowStep_a = (size_t)32 * K;

  f32x8 acc[2][4];
#pragma unroll
  for (int i = 0; i < 2; ++i)
#pragma unroll
    for (int j = 0; j < 4; ++j)
#pragma unroll
      for (int r = 0; r < 8; ++r) acc[i][j][r] = 0.0f;

  // issue one stage (8 async transfers) into buffer `buf` for k-offset k0
  auto stage = [&](int buf, int k0) {
#pragma unroll
    for (int p = 0; p < 4; ++p) {
      async_g2l_b128(lA0 + buf * bufStride + p * rowStep, gA + p * gRowStep_a + k0);
      async_g2l_b128(lB0 + buf * bufStride + p * rowStep, gB + p * gRowStep_a + k0);
    }
  };

  stage(0, 0);
  int cur = 0;
  for (int k0 = 0; k0 < K; k0 += 64) {
    if (k0 + 64 < K) {
      stage(cur ^ 1, k0 + 64);
      if (k0 + 128 < K) {  // warm L2 for the stage after next
        __builtin_prefetch(gA + k0 + 128, 0, 1);
        __builtin_prefetch(gB + k0 + 128, 0, 1);
      }
      asm volatile("s_wait_asynccnt 0x8" ::: "memory");
    } else {
      asm volatile("s_wait_asynccnt 0x0" ::: "memory");
    }
    __syncthreads();

    const bf16* cA = &tA[cur][0];
    const bf16* cB = &tB[cur][0];
#pragma unroll
    for (int kk = 0; kk < 64; kk += 32) {
      bf16x16 a0 = load_frag_rows(cA, 64, wm * 32 + mn,      kk, half);
      bf16x16 a1 = load_frag_rows(cA, 64, wm * 32 + 16 + mn, kk, half);
#pragma unroll
      for (int j = 0; j < 4; ++j) {
        bf16x16 bj = load_frag_rows(cB, 64, wn * 64 + j * 16 + mn, kk, half);
        acc[0][j] = wmma_bf16(a0, bj, acc[0][j]);
        acc[1][j] = wmma_bf16(a1, bj, acc[1][j]);
      }
    }

    __syncthreads();
    cur ^= 1;
  }

#pragma unroll
  for (int i = 0; i < 2; ++i) {
#pragma unroll
    for (int j = 0; j < 4; ++j) {
#pragma unroll
      for (int r = 0; r < 8; ++r) {
        int row = mbase + wm * 32 + i * 16 + r + 8 * half;
        int col = nbase + wn * 64 + j * 16 + mn;
        size_t idx = (size_t)row * N + col;
        float v = acc[i][j][r] + bias[col];
        if (MODE == 1) {
          outF[idx] = v + residual[idx];
        } else if (MODE == 2) {
          float g = 0.5f * v * (1.0f + erff(v * 0.70710678118654752f));
          outB[idx] = f2bf(g);
        } else if (MODE == 3) {
          int bb = row / S, s = row % S;
          outB[((size_t)bb * N + col) * S + s] = f2bf(v);
        } else {
          outB[idx] = f2bf(v);
        }
      }
    }
  }
}

// ---------- flash attention: causal, one wave per 16-query tile per (b,h) ----
// V is pre-transposed: VT[b][col(=h*dh+d)][s]  -> contiguous fragment loads.

__global__ void attention_kernel(const bf16* __restrict__ Q, const bf16* __restrict__ Km,
                                 const bf16* __restrict__ VT, bf16* __restrict__ O,
                                 int Bsz, int S, int H, int dh, int D) {
  __shared__ __align__(16) bf16 pshm[8][16 * 32];   // per-wave P staging tile

  int lane = threadIdx.x & 31;
  int half = lane >> 4;
  int mn   = lane & 15;
  int w    = threadIdx.x >> 5;
  int wid  = (int)((blockIdx.x * blockDim.x + threadIdx.x) >> 5);

  int qtiles = S >> 4;
  int qt = wid % qtiles;
  int h  = (wid / qtiles) % H;
  int b  = wid / (qtiles * H);
  if (b >= Bsz) return;
  int qb = qt * 16;

  const bf16* Qh  = Q  + (size_t)b * S * D + (size_t)h * dh;
  const bf16* Kh  = Km + (size_t)b * S * D + (size_t)h * dh;
  const bf16* VTh = VT + ((size_t)b * D + (size_t)h * dh) * S;  // rows: d, ld = S

  bf16x16 qf0 = load_frag_rows(Qh, D, qb + mn, 0, half);
  bf16x16 qf1 = load_frag_rows(Qh, D, qb + mn, 32, half);

  f32x8 o[4];
#pragma unroll
  for (int t = 0; t < 4; ++t)
#pragma unroll
    for (int i = 0; i < 8; ++i) o[t][i] = 0.0f;
  float m[8], l[8];
#pragma unroll
  for (int r = 0; r < 8; ++r) { m[r] = -1e30f; l[r] = 0.0f; }

  const float scale = 0.125f;   // 1/sqrt(64)

  for (int j0 = 0; j0 <= qb + 15; j0 += 32) {
    f32x8 s0, s1;
#pragma unroll
    for (int i = 0; i < 8; ++i) { s0[i] = 0.0f; s1[i] = 0.0f; }
    {
      bf16x16 kf = load_frag_rows(Kh, D, j0 + mn, 0, half);
      s0 = wmma_bf16(qf0, kf, s0);
      kf = load_frag_rows(Kh, D, j0 + mn, 32, half);
      s0 = wmma_bf16(qf1, kf, s0);
      kf = load_frag_rows(Kh, D, j0 + 16 + mn, 0, half);
      s1 = wmma_bf16(qf0, kf, s1);
      kf = load_frag_rows(Kh, D, j0 + 16 + mn, 32, half);
      s1 = wmma_bf16(qf1, kf, s1);
    }
    // scale + causal mask (C layout: row = r + 8*half, col = mn)
#pragma unroll
    for (int r = 0; r < 8; ++r) {
      int qrow = qb + r + 8 * half;
      s0[r] = (j0 + mn      <= qrow) ? s0[r] * scale : -1e30f;
      s1[r] = (j0 + 16 + mn <= qrow) ? s1[r] * scale : -1e30f;
    }
    // online softmax update
#pragma unroll
    for (int r = 0; r < 8; ++r) {
      float mx = fmaxf(s0[r], s1[r]);
#pragma unroll
      for (int msk = 1; msk < 16; msk <<= 1)
        mx = fmaxf(mx, __shfl_xor(mx, msk, 32));
      float mnew = fmaxf(m[r], mx);
      float alpha = __expf(m[r] - mnew);
      m[r] = mnew;
      float p0 = __expf(s0[r] - mnew);
      float p1 = __expf(s1[r] - mnew);
      s0[r] = p0; s1[r] = p1;
      float rs = p0 + p1;
#pragma unroll
      for (int msk = 1; msk < 16; msk <<= 1)
        rs += __shfl_xor(rs, msk, 32);
      l[r] = l[r] * alpha + rs;
#pragma unroll
      for (int t = 0; t < 4; ++t) o[t][r] *= alpha;
    }
    // repack P (16x32 bf16) through LDS into A-fragment layout
#pragma unroll
    for (int r = 0; r < 8; ++r) {
      int row = r + 8 * half;
      pshm[w][row * 32 + mn]      = f2bf(s0[r]);
      pshm[w][row * 32 + 16 + mn] = f2bf(s1[r]);
    }
    asm volatile("s_wait_dscnt 0x0" ::: "memory");
    bf16x16 pf = load_frag_rows(&pshm[w][0], 32, mn, 0, half);
    // O += P @ V(j0..j0+31, 0..63)  -- V fragments from transposed VT
#pragma unroll
    for (int t = 0; t < 4; ++t) {
      bf16x16 vf = load_frag_rows(VTh, S, t * 16 + mn, j0, half);
      o[t] = wmma_bf16(pf, vf, o[t]);
    }
  }

#pragma unroll
  for (int r = 0; r < 8; ++r) {
    float inv = 1.0f / l[r];
    int row = qb + r + 8 * half;
#pragma unroll
    for (int t = 0; t < 4; ++t) {
      size_t idx = ((size_t)b * S + row) * D + h * dh + t * 16 + mn;
      O[idx] = f2bf(o[t][r] * inv);
    }
  }
}

// ---------- layernorm (block per token) ----------

__global__ void layernorm_kernel(const float* __restrict__ in,
                                 const float* __restrict__ g,
                                 const float* __restrict__ be,
                                 float* __restrict__ outF, bf16* __restrict__ outB,
                                 int D) {
  __shared__ float red[256];
  int row = blockIdx.x;
  int tid = threadIdx.x;
  const float* x = in + (size_t)row * D;

  float s = 0.0f;
  for (int i = tid; i < D; i += 256) s += x[i];
  red[tid] = s; __syncthreads();
  for (int off = 128; off > 0; off >>= 1) {
    if (tid < off) red[tid] += red[tid + off];
    __syncthreads();
  }
  float mu = red[0] / (float)D;
  __syncthreads();

  float v = 0.0f;
  for (int i = tid; i < D; i += 256) { float d = x[i] - mu; v += d * d; }
  red[tid] = v; __syncthreads();
  for (int off = 128; off > 0; off >>= 1) {
    if (tid < off) red[tid] += red[tid + off];
    __syncthreads();
  }
  float rstd = rsqrtf(red[0] / (float)D + 1e-6f);

  for (int i = tid; i < D; i += 256) {
    float y = (x[i] - mu) * rstd * g[i] + be[i];
    outF[(size_t)row * D + i] = y;
    if (outB) outB[(size_t)row * D + i] = f2bf(y);
  }
}

// ---------- host ----------

extern "C" void kernel_launch(void* const* d_in, const int* in_sizes, int n_in,
                              void* d_out, int out_size, void* d_ws, size_t ws_size,
                              hipStream_t stream) {
  const int B = 2, S = 2048, D = 1024, H = 16, dh = 64, F = 4096;
  const long NT = (long)B * S;          // 4096 tokens

  const float* x     = (const float*)d_in[0];
  const float* wq    = (const float*)d_in[1];
  const float* bq    = (const float*)d_in[2];
  const float* wk    = (const float*)d_in[3];
  const float* bk    = (const float*)d_in[4];
  const float* wv    = (const float*)d_in[5];
  const float* bv    = (const float*)d_in[6];
  const float* wo    = (const float*)d_in[7];
  const float* bo    = (const float*)d_in[8];
  const float* wff1  = (const float*)d_in[9];
  const float* bff1  = (const float*)d_in[10];
  const float* wff2  = (const float*)d_in[11];
  const float* bff2  = (const float*)d_in[12];
  const float* ln1g  = (const float*)d_in[13];
  const float* ln1b  = (const float*)d_in[14];
  const float* ln2g  = (const float*)d_in[15];
  const float* ln2b  = (const float*)d_in[16];

  char* ws = (char*)d_ws;
  size_t off = 0;
  auto carve = [&](size_t bytes) -> void* {
    void* p = ws + off;
    off = (off + bytes + 255) & ~(size_t)255;
    return p;
  };

  bf16* xb   = (bf16*)carve(NT * D * 2);
  bf16* wqT  = (bf16*)carve((size_t)D * D * 2);
  bf16* wkT  = (bf16*)carve((size_t)D * D * 2);
  bf16* wvT  = (bf16*)carve((size_t)D * D * 2);
  bf16* woT  = (bf16*)carve((size_t)D * D * 2);
  bf16* w1T  = (bf16*)carve((size_t)D * F * 2);   // F x D
  bf16* w2T  = (bf16*)carve((size_t)F * D * 2);   // D x F
  bf16* qb_  = (bf16*)carve(NT * D * 2);
  bf16* kb_  = (bf16*)carve(NT * D * 2);
  bf16* vT_  = (bf16*)carve(NT * D * 2);          // [b][col][s]
  bf16* ab_  = (bf16*)carve(NT * D * 2);
  float* pre1 = (float*)carve(NT * D * 4);
  float* o1f  = (float*)carve(NT * D * 4);
  bf16*  o1b  = (bf16*)carve(NT * D * 2);
  bf16*  hb   = (bf16*)carve(NT * F * 2);
  float* pre2 = (float*)carve(NT * D * 4);
  (void)ws_size;

  // 1) conversions
  {
    long n = NT * D;
    cvt_bf16_kernel<<<(int)((n + 255) / 256), 256, 0, stream>>>(x, xb, n);
    long nw = (long)D * D;
    cvt_bf16_T_kernel<<<(int)((nw + 255) / 256), 256, 0, stream>>>(wq, wqT, D, D);
    cvt_bf16_T_kernel<<<(int)((nw + 255) / 256), 256, 0, stream>>>(wk, wkT, D, D);
    cvt_bf16_T_kernel<<<(int)((nw + 255) / 256), 256, 0, stream>>>(wv, wvT, D, D);
    cvt_bf16_T_kernel<<<(int)((nw + 255) / 256), 256, 0, stream>>>(wo, woT, D, D);
    long nf = (long)D * F;
    cvt_bf16_T_kernel<<<(int)((nf + 255) / 256), 256, 0, stream>>>(wff1, w1T, D, F);
    cvt_bf16_T_kernel<<<(int)((nf + 255) / 256), 256, 0, stream>>>(wff2, w2T, F, D);
  }

  // 2) QKV projections: (NT x D) @ (D x D) + bias
  {
    int M = (int)NT, N = D, K = D;
    int blocks = (M / 128) * (N / 128);
    gemm_bf16_kernel<0><<<blocks, 256, 0, stream>>>(xb, wqT, bq, nullptr, nullptr, qb_, M, N, K, S);
    gemm_bf16_kernel<0><<<blocks, 256, 0, stream>>>(xb, wkT, bk, nullptr, nullptr, kb_, M, N, K, S);
    gemm_bf16_kernel<3><<<blocks, 256, 0, stream>>>(xb, wvT, bv, nullptr, nullptr, vT_, M, N, K, S);
  }

  // 3) causal flash attention -> ab_ (bf16)
  {
    int waves = B * H * (S / 16);     // 4096
    attention_kernel<<<waves / 8, 256, 0, stream>>>(qb_, kb_, vT_, ab_, B, S, H, dh, D);
  }

  // 4) output projection + residual -> pre1 (fp32)
  {
    int M = (int)NT, N = D, K = D;
    int blocks = (M / 128) * (N / 128);
    gemm_bf16_kernel<1><<<blocks, 256, 0, stream>>>(ab_, woT, bo, x, pre1, nullptr, M, N, K, S);
  }

  // 5) LN1 -> o1f (fp32) + o1b (bf16)
  layernorm_kernel<<<(int)NT, 256, 0, stream>>>(pre1, ln1g, ln1b, o1f, o1b, D);

  // 6) FFN1 + exact GELU -> hb (bf16): (NT x D) @ (D x F)
  {
    int M = (int)NT, N = F, K = D;
    int blocks = (M / 128) * (N / 128);
    gemm_bf16_kernel<2><<<blocks, 256, 0, stream>>>(o1b, w1T, bff1, nullptr, nullptr, hb, M, N, K, S);
  }

  // 7) FFN2 + residual -> pre2 (fp32): (NT x F) @ (F x D)
  {
    int M = (int)NT, N = D, K = F;
    int blocks = (M / 128) * (N / 128);
    gemm_bf16_kernel<1><<<blocks, 256, 0, stream>>>(hb, w2T, bff2, o1f, pre2, nullptr, M, N, K, S);
  }

  // 8) LN2 -> d_out (fp32)
  layernorm_kernel<<<(int)NT, 256, 0, stream>>>(pre2, ln2g, ln2b, (float*)d_out, nullptr, D);
}
